// FocalLoss_75634374082831
// MI455X (gfx1250) — compile-verified, matrix operand order checked by reading, not run on hardware
//
#include <hip/hip_runtime.h>
#include <hip/hip_bf16.h>
#include <math.h>

#define BB 32
#define AA 32768
#define CC 21

typedef __attribute__((ext_vector_type(16))) _Float16 v16h;
typedef __attribute__((ext_vector_type(8)))  float    v8f;

// ---------- order-preserving float<->uint key (ascending key == ascending float) ----------
__device__ __forceinline__ unsigned orderKey(float f) {
  unsigned u = __float_as_uint(f);
  return (u & 0x80000000u) ? ~u : (u | 0x80000000u);
}
__device__ __forceinline__ float keyToFloat(unsigned k) {
  unsigned u = (k & 0x80000000u) ? (k & 0x7FFFFFFFu) : ~k;
  return __uint_as_float(u);
}

// ---------- K0: zero scratch counters/histograms ----------
__global__ void k_zero(unsigned* __restrict__ p, int n) {
  int i = blockIdx.x * blockDim.x + threadIdx.x;
  if (i < n) p[i] = 0u;
}

// ---------- K1: main streaming pass ----------
// Per wave, tiles of 16 anchors. Lane pair (l, l+16) holds anchor l's 21 classes in the
// exact ISA 16-bit A-fragment layout; one v_wmma_f32_16x16x32_f16 with B=ones gives
// sum(exp(x - max)) per anchor (rows). Handler lanes (0-7, 16-23) finish CE, smooth-L1,
// positives count, CE store, and the level-1 (top 11 bits) LDS histogram.
__global__ __launch_bounds__(256)
void k_main(const float* __restrict__ loc_p, const float* __restrict__ loc_t,
            const float* __restrict__ cls_p, const int* __restrict__ cls_t,
            float* __restrict__ ce_out, unsigned* __restrict__ hist1,
            unsigned* __restrict__ num_pos,
            float* __restrict__ part_loc, float* __restrict__ part_posce)
{
  __shared__ unsigned lh[2048];
  __shared__ float    fred[256];
  __shared__ unsigned ured[256];

  const int CHUNKS = AA / 2048;          // 16 blocks per image
  int b = blockIdx.x / CHUNKS;
  int chunk = blockIdx.x % CHUNKS;
  int tid = threadIdx.x;
  int lane = tid & 31;
  int wave = tid >> 5;

  for (int i = tid; i < 2048; i += 256) lh[i] = 0u;
  __syncthreads();

  v16h onesB;
#pragma unroll
  for (int i = 0; i < 16; ++i) onesB[i] = (_Float16)1.0f;

  float loc_acc = 0.f, posce_acc = 0.f;
  unsigned cnt_acc = 0u;
  bool loLane = (lane < 16);
  int rsel = lane & 7;
  bool handler = ((lane & 8) == 0);      // lanes 0-7 -> rows 0-7, lanes 16-23 -> rows 8-15
  int row = loLane ? rsel : (8 + rsel);

  for (int t = 0; t < 16; ++t) {
    int tileBase = b * AA + chunk * 2048 + (wave * 16 + t) * 16;  // flat anchor index base
    int myAnchor = tileBase + (lane & 15);
    const float* rowp = cls_p + (size_t)myAnchor * CC;

    float x[13];
    if (loLane) {
#pragma unroll
      for (int j = 0; j < 8; ++j) x[j] = rowp[j];          // K 0..7
#pragma unroll
      for (int j = 0; j < 5; ++j) x[8 + j] = rowp[16 + j]; // K 16..20
    } else {
#pragma unroll
      for (int j = 0; j < 8; ++j) x[j] = rowp[8 + j];      // K 8..15
#pragma unroll
      for (int j = 0; j < 5; ++j) x[8 + j] = -INFINITY;    // K 24.. pad
    }
    float mx = x[0];
#pragma unroll
    for (int j = 1; j < 13; ++j) mx = fmaxf(mx, x[j]);
    mx = fmaxf(mx, __shfl_xor(mx, 16, 32));                // full per-anchor max

    v16h afrag;
#pragma unroll
    for (int j = 0; j < 13; ++j) afrag[j] = (_Float16)__expf(x[j] - mx);
#pragma unroll
    for (int j = 13; j < 16; ++j) afrag[j] = (_Float16)0.f;

    v8f c;
#pragma unroll
    for (int i = 0; i < 8; ++i) c[i] = 0.f;
    c = __builtin_amdgcn_wmma_f32_16x16x32_f16(false, afrag, false, onesB,
                                               (short)0, c, false, false);
    // row sums: lanes<16 hold rows 0-7 (VGPR r), lanes>=16 hold rows 8-15
    float se = 0.f;
#pragma unroll
    for (int r = 0; r < 8; ++r) se = (rsel == r) ? c[r] : se;
    float mrow = __shfl(mx, row, 32);                      // max of my handled row

    if (handler) {
      int am = tileBase + row;
      int tlab = cls_t[am];
      int tgt = (tlab < 0) ? 0 : tlab;
      float picked = cls_p[(size_t)am * CC + tgt];
      float ce = mrow + __logf(se) - picked;
      ce_out[am] = ce;
      unsigned key = orderKey(ce);
      atomicAdd(&lh[key >> 21], 1u);
      if (tlab > 0) {
        cnt_acc++;
        posce_acc += ce;
        const float* lp = loc_p + (size_t)am * 4;
        const float* lt = loc_t + (size_t)am * 4;
#pragma unroll
        for (int j = 0; j < 4; ++j) {
          float d = fabsf(lp[j] - lt[j]);
          loc_acc += (d < 1.0f) ? 0.5f * d * d : (d - 0.5f);
        }
      }
    }
  }

  // block reductions (fixed-order -> deterministic floats; integer atomic for counts)
  fred[tid] = loc_acc; ured[tid] = cnt_acc;
  __syncthreads();
  for (int s = 128; s > 0; s >>= 1) {
    if (tid < s) { fred[tid] += fred[tid + s]; ured[tid] += ured[tid + s]; }
    __syncthreads();
  }
  if (tid == 0) { part_loc[blockIdx.x] = fred[0]; atomicAdd(&num_pos[b], ured[0]); }
  __syncthreads();
  fred[tid] = posce_acc;
  __syncthreads();
  for (int s = 128; s > 0; s >>= 1) {
    if (tid < s) fred[tid] += fred[tid + s];
    __syncthreads();
  }
  if (tid == 0) part_posce[blockIdx.x] = fred[0];
  __syncthreads();
  for (int i = tid; i < 2048; i += 256) {
    unsigned v = lh[i];
    if (v) atomicAdd(&hist1[b * 2048 + i], v);
  }
}

// ---------- deterministic partial-sum reduction ----------
__global__ __launch_bounds__(256)
void k_reduce(const float* __restrict__ in, int n, float* __restrict__ out) {
  __shared__ float s[256];
  float acc = 0.f;
  for (int i = threadIdx.x; i < n; i += 256) acc += in[i];
  s[threadIdx.x] = acc;
  __syncthreads();
  for (int st = 128; st > 0; st >>= 1) {
    if (threadIdx.x < st) s[threadIdx.x] += s[threadIdx.x + st];
    __syncthreads();
  }
  if (threadIdx.x == 0) out[0] = s[0];
}

// ---------- radix-select scan stage (one block per image) ----------
// Finds, scanning bins from the top, the bin where the descending cumulative count
// crosses k. Emits combined prefix (pfx<<lvl_bits | bin) and remaining need.
__global__ __launch_bounds__(256)
void k_scan(const unsigned* __restrict__ hist, int nbins,
            const unsigned* __restrict__ k_in, int k_mode,
            const unsigned* __restrict__ pfx_in, int use_pfx, int lvl_bits,
            unsigned* __restrict__ pfx_out, unsigned* __restrict__ need_out)
{
  __shared__ unsigned ps[256];
  int b = blockIdx.x, tid = threadIdx.x;
  unsigned k;
  if (k_mode == 0) { unsigned kk = 3u * k_in[b]; k = (kk > (unsigned)AA) ? (unsigned)AA : kk; }
  else k = k_in[b];
  unsigned pfx = use_pfx ? pfx_in[b] : 0u;
  bool dead = (k == 0u) || (pfx == 0xFFFFFFFFu);
  if (dead) {
    if (tid == 0) { pfx_out[b] = 0xFFFFFFFFu; need_out[b] = 0u; }
    return;
  }
  int G = nbins >> 8;                     // bins per thread (8 or 4)
  unsigned gsum = 0u;
  for (int j = 0; j < G; ++j) gsum += hist[b * nbins + (nbins - 1 - (tid * G + j))];
  ps[tid] = gsum;
  __syncthreads();
  for (int off = 1; off < 256; off <<= 1) {
    unsigned add = (tid >= off) ? ps[tid - off] : 0u;
    __syncthreads();
    ps[tid] += add;
    __syncthreads();
  }
  unsigned cum = ps[tid] - gsum;          // exclusive prefix (descending-rank order)
  for (int j = 0; j < G; ++j) {
    int bin = nbins - 1 - (tid * G + j);
    unsigned h = hist[b * nbins + bin];
    cum += h;
    if (cum >= k && (cum - h) < k) {      // unique crossing thread
      pfx_out[b]  = (pfx << lvl_bits) | (unsigned)bin;
      need_out[b] = k - (cum - h);
    }
  }
}

// ---------- refinement histogram over CE (levels 2 and 3) ----------
__global__ __launch_bounds__(256)
void k_hist(const float* __restrict__ ce, const unsigned* __restrict__ matchVal,
            unsigned* __restrict__ hist, int nbins, int matchShift, int binShift)
{
  __shared__ unsigned lh[2048];
  const int CHUNKS = AA / 4096;           // 8 blocks per image
  int b = blockIdx.x / CHUNKS;
  int chunk = blockIdx.x % CHUNKS;
  int tid = threadIdx.x;
  unsigned mv = matchVal[b];
  for (int i = tid; i < nbins; i += 256) lh[i] = 0u;
  __syncthreads();
  int base = b * AA + chunk * 4096;
  for (int i = tid; i < 4096; i += 256) {
    unsigned key = orderKey(ce[base + i]);
    if ((key >> matchShift) == mv) atomicAdd(&lh[(key >> binShift) & (nbins - 1)], 1u);
  }
  __syncthreads();
  for (int i = tid; i < nbins; i += 256) {
    unsigned v = lh[i];
    if (v) atomicAdd(&hist[b * nbins + i], v);
  }
}

// ---------- final top-k sum: WMMA accumulator, 512 terms per instruction ----------
__global__ __launch_bounds__(256)
void k_topk(const float* __restrict__ ce, const unsigned* __restrict__ thr,
            float* __restrict__ part_topk)
{
  int tid0 = blockIdx.x * 256 + threadIdx.x;   // 65536 threads total
  const int STR = 65536;
  v16h a, onesB;
  v8f c;
#pragma unroll
  for (int i = 0; i < 16; ++i) onesB[i] = (_Float16)1.0f;
#pragma unroll
  for (int i = 0; i < 8; ++i) c[i] = 0.f;
#pragma unroll
  for (int j = 0; j < 16; ++j) {
    int idx = tid0 + j * STR;                  // covers all B*A = 1,048,576
    int b = idx >> 15;                         // AA = 2^15
    float v = ce[idx];
    unsigned key = orderKey(v);
    a[j] = (_Float16)((key > thr[b]) ? v : 0.f);
  }
  c = __builtin_amdgcn_wmma_f32_16x16x32_f16(false, a, false, onesB,
                                             (short)0, c, false, false);
  float s = 0.f;
#pragma unroll
  for (int i = 0; i < 8; ++i) s += c[i];
  // D columns replicate row sums: column 0 lives in lane 0 (rows 0-7) + lane 16 (rows 8-15)
  float tot = __shfl(s, 0, 32) + __shfl(s, 16, 32);
  int lane = threadIdx.x & 31, wave = threadIdx.x >> 5;
  if (lane == 0) part_topk[blockIdx.x * 8 + wave] = tot;
}

// ---------- combine ----------
__global__ void k_final(const unsigned* __restrict__ num_pos, const unsigned* __restrict__ thr,
                        const unsigned* __restrict__ rcnt, const float* __restrict__ red,
                        float* __restrict__ out)
{
  if (blockIdx.x == 0 && threadIdx.x == 0) {
    float extra = 0.f;
    unsigned N = 0u;
    for (int b = 0; b < BB; ++b) {
      N += num_pos[b];
      unsigned r = rcnt[b];
      if (r) extra += (float)r * keyToFloat(thr[b]);   // ties at threshold: all equal
    }
    out[0] = (red[0] + red[1] + red[2] + extra) / (float)N;
  }
}

extern "C" void kernel_launch(void* const* d_in, const int* in_sizes, int n_in,
                              void* d_out, int out_size, void* d_ws, size_t ws_size,
                              hipStream_t stream) {
  const float* loc_p = (const float*)d_in[0];
  const float* loc_t = (const float*)d_in[1];
  const float* cls_p = (const float*)d_in[2];
  const int*   cls_t = (const int*)d_in[3];
  float* out = (float*)d_out;
  char* ws = (char*)d_ws;

  float*    ce      = (float*)ws;                         // 4,194,304 B
  unsigned* hist1   = (unsigned*)(ws + 4194304);          // 262,144 B
  unsigned* hist2   = (unsigned*)(ws + 4456448);          // 262,144 B
  unsigned* hist3   = (unsigned*)(ws + 4718592);          // 131,072 B
  unsigned* num_pos = (unsigned*)(ws + 4849664);          // 32
  unsigned* pfx1  = num_pos + 32;
  unsigned* need1 = pfx1 + 32;
  unsigned* pfx2  = need1 + 32;
  unsigned* need2 = pfx2 + 32;
  unsigned* thr   = need2 + 32;                           // final: exact threshold key
  unsigned* rcnt  = thr + 32;                             // final: #ties to include
  float* part_loc   = (float*)(rcnt + 32);                // 512
  float* part_posce = part_loc + 512;                     // 512
  float* part_topk  = part_posce + 512;                   // 2048
  float* red        = part_topk + 2048;                   // [loc, posce, topk]

  // zero hist1..hist3 + num_pos (everything atomically accumulated)
  const int ZN = (262144 * 2 + 131072 + 128) / 4;         // 163,872 u32
  k_zero<<<(ZN + 255) / 256, 256, 0, stream>>>((unsigned*)(ws + 4194304), ZN);

  k_main<<<BB * (AA / 2048), 256, 0, stream>>>(loc_p, loc_t, cls_p, cls_t,
                                               ce, hist1, num_pos, part_loc, part_posce);
  k_reduce<<<1, 256, 0, stream>>>(part_loc, 512, red + 0);
  k_reduce<<<1, 256, 0, stream>>>(part_posce, 512, red + 1);

  // 3-level exact radix select per image: 11 + 11 + 10 bits of the ordered key
  k_scan<<<BB, 256, 0, stream>>>(hist1, 2048, num_pos, 0, num_pos, 0, 11, pfx1, need1);
  k_hist<<<BB * (AA / 4096), 256, 0, stream>>>(ce, pfx1, hist2, 2048, 21, 10);
  k_scan<<<BB, 256, 0, stream>>>(hist2, 2048, need1, 1, pfx1, 1, 11, pfx2, need2);
  k_hist<<<BB * (AA / 4096), 256, 0, stream>>>(ce, pfx2, hist3, 1024, 10, 0);
  k_scan<<<BB, 256, 0, stream>>>(hist3, 1024, need2, 1, pfx2, 1, 10, thr, rcnt);

  k_topk<<<256, 256, 0, stream>>>(ce, thr, part_topk);
  k_reduce<<<1, 256, 0, stream>>>(part_topk, 2048, red + 2);
  k_final<<<1, 32, 0, stream>>>(num_pos, thr, rcnt, red, out);
}
